// MultiMemoryAttention_28054726377483
// MI455X (gfx1250) — matrix operand order repaired and checked by reading, not compiled, not run on hardware
//
#include <hip/hip_runtime.h>
#include <hip/hip_bf16.h>
#include <stdint.h>

// ---------------- types ----------------
typedef __attribute__((ext_vector_type(2)))  float  v2f;
typedef __attribute__((ext_vector_type(8)))  float  v8f;
typedef __attribute__((ext_vector_type(16))) __bf16 v16bf;

#define HNUM 16
#define DNUM 64
#define SNUM 4096
#define BNUM 2
#define HID  1024
#define MNUM 4
#define LSEQ 256
#define TTOK 257
#define CHK  128
#define EPS_F 1e-6f

// ---------------- scalar helpers ----------------
__device__ __forceinline__ unsigned short f32_to_bf16r(float f) {
    unsigned int u = __float_as_uint(f);
    unsigned int r = (u + 0x7FFFu + ((u >> 16) & 1u)) >> 16;
    return (unsigned short)r;
}
__device__ __forceinline__ float bf16r_to_f32(unsigned short h) {
    return __uint_as_float(((unsigned int)h) << 16);
}
__device__ __forceinline__ float elu1(float x) { return x > 0.f ? x + 1.f : __expf(x); }
__device__ __forceinline__ float gelu_exact(float x) {
    return 0.5f * x * (1.f + erff(x * 0.70710678118654752f));
}
__device__ __forceinline__ float wred_sum(float v) {
    #pragma unroll
    for (int off = 16; off > 0; off >>= 1) v += __shfl_xor(v, off, 32);
    return v;
}
__device__ __forceinline__ float wred_max(float v) {
    #pragma unroll
    for (int off = 16; off > 0; off >>= 1) v = fmaxf(v, __shfl_xor(v, off, 32));
    return v;
}

// ---------------- CDNA5 async global->LDS copy (ASYNCcnt-tracked) ----------------
__device__ __forceinline__ void async_copy_b128(void* lds_dst, const void* gsrc) {
    unsigned int la = (unsigned int)(uintptr_t)lds_dst;              // LDS offset = low 32 bits
    unsigned long long ga = (unsigned long long)(uintptr_t)gsrc;
    asm volatile("global_load_async_to_lds_b128 %0, %1, off"
                 :: "v"(la), "v"(ga) : "memory");
}
__device__ __forceinline__ void wait_async0() {
    asm volatile("s_wait_asynccnt 0" ::: "memory");
}

// ---------------- WMMA fragment helpers (wave32 layouts per CDNA5 ISA 7.12.2) ----------------
__device__ __forceinline__ v2f load_a_f32(const float* p, int sM, int sK, int t, int lane) {
    int row = lane & 15;
    int kb  = t * 4 + ((lane & 16) ? 2 : 0);
    v2f a; a.x = p[row * sM + kb * sK]; a.y = p[row * sM + (kb + 1) * sK];
    return a;
}
__device__ __forceinline__ v2f load_b_f32(const float* p, int sK, int sN, int t, int lane) {
    int n  = lane & 15;
    int kb = t * 4 + ((lane & 16) ? 2 : 0);
    v2f b; b.x = p[kb * sK + n * sN]; b.y = p[(kb + 1) * sK + n * sN];
    return b;
}
__device__ __forceinline__ v2f load_a_bf16s(const unsigned short* p, int sM, int sK, int t, int lane) {
    int row = lane & 15;
    int kb  = t * 4 + ((lane & 16) ? 2 : 0);
    v2f a; a.x = bf16r_to_f32(p[row * sM + kb * sK]); a.y = bf16r_to_f32(p[row * sM + (kb + 1) * sK]);
    return a;
}
__device__ __forceinline__ v2f load_b_bf16s(const unsigned short* p, int sK, int sN, int t, int lane) {
    int n  = lane & 15;
    int kb = t * 4 + ((lane & 16) ? 2 : 0);
    v2f b; b.x = bf16r_to_f32(p[kb * sK + n * sN]); b.y = bf16r_to_f32(p[(kb + 1) * sK + n * sN]);
    return b;
}
__device__ __forceinline__ v8f wmma_f32_k4(v2f a, v2f b, v8f c) {
    return __builtin_amdgcn_wmma_f32_16x16x4_f32(false, a, false, b, (short)0, c, false, false);
}

// bf16 WMMA 16x16x32 fragments from row-major bf16 (raw ushort) memory.
__device__ __forceinline__ v16bf load_amat_bf16(const unsigned short* p, int lda, int lane) {
    union { v16bf v; unsigned int u[8]; } r;
    int row = lane & 15;
    int kh  = (lane & 16) ? 8 : 0;
    const unsigned short* rp = p + (size_t)row * lda;
    #pragma unroll
    for (int i = 0; i < 8; ++i) {
        int k = (i >> 2) * 16 + kh + (i & 3) * 2;
        r.u[i] = *(const unsigned int*)(rp + k);
    }
    return r.v;
}
__device__ __forceinline__ v16bf load_bmat_bf16_nt(const unsigned short* p, int ldb, int lane) {
    union { v16bf v; unsigned int u[8]; } r;
    int n    = lane & 15;
    int koff = (lane & 16) ? 8 : 0;
    const unsigned short* np = p + (size_t)n * ldb;
    #pragma unroll
    for (int i = 0; i < 8; ++i) r.u[i] = *(const unsigned int*)(np + koff + (i >> 2) * 16 + (i & 3) * 2);
    return r.v;
}

// ---------------- kernel 0: f32 -> bf16 convert ----------------
__global__ void cvt_f32_bf16_kernel(const float* __restrict__ in, unsigned short* __restrict__ out, long n) {
    long i = (long)blockIdx.x * blockDim.x + threadIdx.x;
    long stride = (long)gridDim.x * blockDim.x;
    for (; i < n; i += stride) out[i] = f32_to_bf16r(in[i]);
}

// ---------------- kernel 1: tiled bf16 WMMA GEMM  C = A @ W^T ----------------
// One wave computes a 16(M) x 64(N) tile: A fragment reused across 4 B fragments.
// mode 0: C row-major (M,N) f32. mode 1: scatter to (B,H,S,D) with M=B*S, N=H*D.
__global__ void gemm_bf16_kernel(const unsigned short* __restrict__ A,
                                 const unsigned short* __restrict__ W,
                                 float* __restrict__ C,
                                 int Mrows, int Nn, int K, int mode) {
    int lane = threadIdx.x & 31;
    int wave = blockIdx.x * 8 + (threadIdx.x >> 5);
    int ntn  = Nn >> 6;                       // tiles of 64 cols
    int tm = wave / ntn, tn = wave % ntn;
    if (tm * 16 >= Mrows) return;
    const unsigned short* Ab = A + (size_t)tm * 16 * K;
    const v8f vzero = {0.f, 0.f, 0.f, 0.f, 0.f, 0.f, 0.f, 0.f};
    v8f acc[4] = {vzero, vzero, vzero, vzero};
    for (int k = 0; k < K; k += 32) {
        v16bf af = load_amat_bf16(Ab + k, K, lane);
        #pragma unroll
        for (int j = 0; j < 4; ++j) {
            const unsigned short* Wb = W + (size_t)(tn * 64 + j * 16) * K;
            v16bf bf = load_bmat_bf16_nt(Wb + k, K, lane);
            acc[j] = __builtin_amdgcn_wmma_f32_16x16x32_bf16(false, af, false, bf, (short)0, acc[j], false, false);
        }
    }
    int col = lane & 15, hi = (lane & 16) ? 8 : 0;
    #pragma unroll
    for (int j = 0; j < 4; ++j)
        #pragma unroll
        for (int i = 0; i < 8; ++i) {
            int r = tm * 16 + i + hi;
            int c = tn * 64 + j * 16 + col;
            if (mode == 0) {
                C[(size_t)r * Nn + c] = acc[j][i];
            } else {
                int b = r >> 12, s = r & (SNUM - 1);
                int h = c >> 6,  d = c & 63;
                C[(((size_t)(b * HNUM + h)) * SNUM + s) * DNUM + d] = acc[j][i];
            }
        }
}

// ---------------- kernel 2: per-(h,m) chunk encoder ----------------
__device__ __forceinline__ float dot64(const float* a, const float* w) {
    float s = 0.f;
    #pragma unroll 8
    for (int i = 0; i < 64; ++i) s += a[i] * w[i];
    return s;
}
__device__ __forceinline__ float dot256(const float* a, const float* w) {
    float s = 0.f;
    #pragma unroll 8
    for (int i = 0; i < 256; ++i) s += a[i] * w[i];
    return s;
}

__global__ void encoder_kernel(const float* __restrict__ key_seqs,
                               const float* __restrict__ cls,
                               const float* __restrict__ in_w, const float* __restrict__ in_b,
                               const float* __restrict__ ao_w, const float* __restrict__ ao_b,
                               const float* __restrict__ ln1_g, const float* __restrict__ ln1_b,
                               const float* __restrict__ ln2_g, const float* __restrict__ ln2_b,
                               const float* __restrict__ ff1_w, const float* __restrict__ ff1_b,
                               const float* __restrict__ ff2_w, const float* __restrict__ ff2_b,
                               const float* __restrict__ proj_w,
                               float* __restrict__ landmarks) {
    __shared__ float xs[TTOK * 64];
    __shared__ unsigned short ksL[TTOK * 64];
    __shared__ unsigned short vsL[TTOK * 64];
    __shared__ float qrow[8][64];
    __shared__ float orow[8][64];
    __shared__ float hrow[8][256];

    int h = blockIdx.x / MNUM;
    int m = blockIdx.x % MNUM;
    int tid = threadIdx.x, wv = tid >> 5, lane = tid & 31;

    const float* seq = key_seqs + ((size_t)(m * HNUM + h)) * LSEQ * 64;
    for (int idx = tid; idx < TTOK * 64; idx += 256) {
        int t = idx >> 6, d = idx & 63;
        xs[idx] = (t == 0) ? cls[h * 64 + d] : seq[(size_t)(t - 1) * 64 + d];
    }
    __syncthreads();

    for (int l = 0; l < 2; ++l) {
        const float* Wi  = in_w  + ((size_t)(h * 2 + l)) * 192 * 64;
        const float* Bi  = in_b  + (size_t)(h * 2 + l) * 192;
        const float* Wao = ao_w  + ((size_t)(h * 2 + l)) * 64 * 64;
        const float* Bao = ao_b  + (size_t)(h * 2 + l) * 64;
        const float* G1  = ln1_g + (size_t)(h * 2 + l) * 64;
        const float* B1  = ln1_b + (size_t)(h * 2 + l) * 64;
        const float* G2  = ln2_g + (size_t)(h * 2 + l) * 64;
        const float* B2  = ln2_b + (size_t)(h * 2 + l) * 64;
        const float* F1w = ff1_w + ((size_t)(h * 2 + l)) * 256 * 64;
        const float* F1b = ff1_b + (size_t)(h * 2 + l) * 256;
        const float* F2w = ff2_w + ((size_t)(h * 2 + l)) * 64 * 256;
        const float* F2b = ff2_b + (size_t)(h * 2 + l) * 64;

        for (int idx = tid; idx < TTOK * 64; idx += 256) {
            int t = idx >> 6, c = idx & 63;
            float kvl = Bi[64 + c]  + dot64(xs + t * 64, Wi + (size_t)(64 + c) * 64);
            float vvl = Bi[128 + c] + dot64(xs + t * 64, Wi + (size_t)(128 + c) * 64);
            ksL[idx] = f32_to_bf16r(kvl);
            vsL[idx] = f32_to_bf16r(vvl);
        }
        __syncthreads();

        for (int t = wv; t < TTOK; t += 8) {
            for (int c = lane; c < 64; c += 32)
                qrow[wv][c] = Bi[c] + dot64(xs + t * 64, Wi + (size_t)c * 64);
            #pragma unroll
            for (int ne = 0; ne < 4; ++ne) {
                int basec = ne * 16;
                float mx = -3.0e30f;
                for (int s = lane; s < TTOK; s += 32) {
                    float sc = 0.f;
                    #pragma unroll
                    for (int j = 0; j < 16; ++j)
                        sc += qrow[wv][basec + j] * bf16r_to_f32(ksL[s * 64 + basec + j]);
                    sc *= 0.25f;
                    mx = fmaxf(mx, sc);
                }
                mx = wred_max(mx);
                float den = 0.f;
                for (int s = lane; s < TTOK; s += 32) {
                    float sc = 0.f;
                    #pragma unroll
                    for (int j = 0; j < 16; ++j)
                        sc += qrow[wv][basec + j] * bf16r_to_f32(ksL[s * 64 + basec + j]);
                    den += __expf(sc * 0.25f - mx);
                }
                den = wred_sum(den);
                if (lane < 16) {
                    float acc = 0.f;
                    for (int s = 0; s < TTOK; ++s) {
                        float sc = 0.f;
                        #pragma unroll
                        for (int j = 0; j < 16; ++j)
                            sc += qrow[wv][basec + j] * bf16r_to_f32(ksL[s * 64 + basec + j]);
                        float a = __expf(sc * 0.25f - mx);
                        acc += a * bf16r_to_f32(vsL[s * 64 + basec + lane]);
                    }
                    orow[wv][basec + lane] = acc / den;
                }
            }
            int c0 = lane, c1 = lane + 32;
            float r0 = xs[t * 64 + c0] + Bao[c0] + dot64(orow[wv], Wao + (size_t)c0 * 64);
            float r1 = xs[t * 64 + c1] + Bao[c1] + dot64(orow[wv], Wao + (size_t)c1 * 64);
            float mu  = wred_sum(r0 + r1) * (1.f / 64.f);
            float mu2 = wred_sum(r0 * r0 + r1 * r1) * (1.f / 64.f);
            float inv = rsqrtf(mu2 - mu * mu + 1e-5f);
            xs[t * 64 + c0] = (r0 - mu) * inv * G1[c0] + B1[c0];
            xs[t * 64 + c1] = (r1 - mu) * inv * G1[c1] + B1[c1];
            for (int c2 = lane; c2 < 256; c2 += 32)
                hrow[wv][c2] = gelu_exact(F1b[c2] + dot64(xs + t * 64, F1w + (size_t)c2 * 64));
            float f0 = xs[t * 64 + c0] + F2b[c0] + dot256(hrow[wv], F2w + (size_t)c0 * 256);
            float f1 = xs[t * 64 + c1] + F2b[c1] + dot256(hrow[wv], F2w + (size_t)c1 * 256);
            mu  = wred_sum(f0 + f1) * (1.f / 64.f);
            mu2 = wred_sum(f0 * f0 + f1 * f1) * (1.f / 64.f);
            inv = rsqrtf(mu2 - mu * mu + 1e-5f);
            xs[t * 64 + c0] = (f0 - mu) * inv * G2[c0] + B2[c0];
            xs[t * 64 + c1] = (f1 - mu) * inv * G2[c1] + B2[c1];
        }
        __syncthreads();
    }
    const float* Pw = proj_w + (size_t)h * 64 * 64;
    if (tid < 64)
        landmarks[((size_t)(m * HNUM + h)) * 64 + tid] = dot64(xs, Pw + (size_t)tid * 64);
}

// ---------------- kernel 3: landmark relevance softmax weights ----------------
__global__ void landmark_wts_kernel(const float* __restrict__ qslc,
                                    const float* __restrict__ lm,
                                    float* __restrict__ wts) {
    __shared__ float red[256 * 4];
    int b = blockIdx.x >> 4, h = blockIdx.x & 15;
    int tid = threadIdx.x;
    float part[4] = {0.f, 0.f, 0.f, 0.f};
    const float* qb = qslc + ((size_t)(b * HNUM + h)) * SNUM * 64;
    for (int s = tid; s < SNUM; s += 256) {
        const float* qs = qb + (size_t)s * 64;
        #pragma unroll
        for (int m = 0; m < 4; ++m) {
            const float* lmv = lm + ((size_t)(m * HNUM + h)) * 64;
            float d = 0.f;
            #pragma unroll 8
            for (int e = 0; e < 64; ++e) d += qs[e] * lmv[e];
            part[m] += d;
        }
    }
    #pragma unroll
    for (int m = 0; m < 4; ++m) red[tid * 4 + m] = part[m];
    __syncthreads();
    for (int st = 128; st > 0; st >>= 1) {
        if (tid < st)
            #pragma unroll
            for (int m = 0; m < 4; ++m) red[tid * 4 + m] += red[(tid + st) * 4 + m];
        __syncthreads();
    }
    if (tid == 0) {
        const float sc = 0.125f / (float)SNUM;
        float r[4], mx = -3e30f;
        #pragma unroll
        for (int m = 0; m < 4; ++m) { r[m] = red[m] * sc; mx = fmaxf(mx, r[m]); }
        float den = 0.f;
        #pragma unroll
        for (int m = 0; m < 4; ++m) { r[m] = __expf(r[m] - mx); den += r[m]; }
        #pragma unroll
        for (int m = 0; m < 4; ++m) wts[m * (BNUM * HNUM) + b * HNUM + h] = r[m] / den;
    }
}

// ---------------- kernel 4: memory readout (f32 WMMA) ----------------
__global__ void mem_readout_kernel(const float* __restrict__ qb,
                                   const float* __restrict__ mems,
                                   const float* __restrict__ norms,
                                   const float* __restrict__ wts,
                                   float* __restrict__ memout) {
    __shared__ float sig[8][16 * 64];
    __shared__ float den[8][4][16];
    int wv = threadIdx.x >> 5, lane = threadIdx.x & 31;
    int gwave = blockIdx.x * 8 + wv;
    int bh = gwave >> 8;
    int ts = gwave & 255;
    int b = bh >> 4, h = bh & 15;
    const float* qt = qb + ((size_t)bh * SNUM + (size_t)ts * 16) * 64;
    float* ssig = sig[wv];
    for (int idx = lane; idx < 16 * 64; idx += 32) ssig[idx] = elu1(qt[idx]);
    if (lane < 16) {
        #pragma unroll
        for (int m = 0; m < 4; ++m) {
            const float* nm = norms + ((size_t)(m * HNUM + h)) * 64;
            float s = 0.f;
            #pragma unroll 8
            for (int d = 0; d < 64; ++d) s += ssig[lane * 64 + d] * nm[d];
            den[wv][m][lane] = s;
        }
    }
    const v8f vzero = {0.f, 0.f, 0.f, 0.f, 0.f, 0.f, 0.f, 0.f};
    v8f acc[4] = {vzero, vzero, vzero, vzero};
    int hi = (lane & 16) ? 8 : 0;
    #pragma unroll
    for (int m = 0; m < 4; ++m) {
        float wm = wts[m * (BNUM * HNUM) + b * HNUM + h];
        float scl[8];
        #pragma unroll
        for (int i = 0; i < 8; ++i) scl[i] = wm / fmaxf(den[wv][m][i + hi], EPS_F);
        const float* Mh = mems + ((size_t)(m * HNUM + h)) * 64 * 64;
        #pragma unroll
        for (int j = 0; j < 4; ++j) {
            v8f nm = vzero;
            for (int t = 0; t < 16; ++t) {
                v2f a  = load_a_f32(ssig, 64, 1, t, lane);
                v2f bb = load_b_f32(Mh + j * 16, 64, 1, t, lane);
                nm = wmma_f32_k4(a, bb, nm);
            }
            #pragma unroll
            for (int i = 0; i < 8; ++i) acc[j][i] += nm[i] * scl[i];
        }
    }
    int col = lane & 15;
    #pragma unroll
    for (int j = 0; j < 4; ++j)
        #pragma unroll
        for (int i = 0; i < 8; ++i)
            memout[((size_t)bh * SNUM + ts * 16 + i + hi) * 64 + j * 16 + col] = acc[j][i];
}

// ---------------- kernel 5: chunked causal linear attention + gate fusion ----------------
__global__ void causal_kernel(const float* __restrict__ qb, const float* __restrict__ kb,
                              const float* __restrict__ vb, const float* __restrict__ memout,
                              const float* __restrict__ gate, unsigned short* __restrict__ Ybf) {
    __shared__ float sq[CHK * 64];
    __shared__ unsigned short skL[CHK * 64];
    __shared__ unsigned short svL[CHK * 64];
    __shared__ float sc[CHK * CHK];          // scores; also raw k/v staging at chunk start
    __shared__ float Skv[64 * 64];
    __shared__ float zz[64];
    __shared__ float den[CHK];

    int bh = blockIdx.x, b = bh >> 4, h = bh & 15;
    int tid = threadIdx.x, wv = tid >> 5, lane = tid & 31;
    int col = lane & 15, hi = (lane & 16) ? 8 : 0;
    float g = 1.f / (1.f + __expf(-gate[h]));
    const size_t base = (size_t)bh * SNUM * 64;
    const v8f vzero = {0.f, 0.f, 0.f, 0.f, 0.f, 0.f, 0.f, 0.f};

    for (int i = tid; i < 64 * 64; i += 256) Skv[i] = 0.f;
    if (tid < 64) zz[tid] = 0.f;
    __syncthreads();

    float* rawk = sc;
    float* rawv = sc + CHK * 64;

    for (int ch = 0; ch < SNUM / CHK; ++ch) {
        int s0 = ch * CHK;
        const float* gq = qb + base + (size_t)s0 * 64;
        const float* gk = kb + base + (size_t)s0 * 64;
        const float* gv = vb + base + (size_t)s0 * 64;
        // async DMA raw chunk: q->sq (in-place transform later), k/v->sc staging
        for (int idx = tid * 4; idx < CHK * 64; idx += 256 * 4) {
            async_copy_b128(sq + idx,   gq + idx);
            async_copy_b128(rawk + idx, gk + idx);
            async_copy_b128(rawv + idx, gv + idx);
        }
        wait_async0();
        __syncthreads();
        // elementwise transforms (in-place for q)
        for (int idx = tid; idx < CHK * 64; idx += 256) {
            sq[idx]  = elu1(sq[idx]);
            skL[idx] = f32_to_bf16r(elu1(rawk[idx]));
            svL[idx] = f32_to_bf16r(rawv[idx]);
        }
        __syncthreads();
        // prefetch next chunk while computing this one
        if (ch + 1 < SNUM / CHK) {
            const float* nq = gq + CHK * 64;
            const float* nk = gk + CHK * 64;
            const float* nv = gv + CHK * 64;
            __builtin_prefetch(nq + tid * 32, 0, 3);
            __builtin_prefetch(nk + tid * 32, 0, 3);
            __builtin_prefetch(nv + tid * 32, 0, 3);
        }

        // cache sigma_q A fragments for this wave's 16 query rows
        const float* aq = sq + wv * 16 * 64;
        v2f afq[16];
        #pragma unroll
        for (int t = 0; t < 16; ++t) afq[t] = load_a_f32(aq, 64, 1, t, lane);

        // phase 1: num = sigma_q @ Skv (old state)
        v8f num[4] = {vzero, vzero, vzero, vzero};
        #pragma unroll
        for (int t = 0; t < 16; ++t) {
            #pragma unroll
            for (int j = 0; j < 4; ++j) {
                v2f bb = load_b_f32(Skv + j * 16, 64, 1, t, lane);
                num[j] = wmma_f32_k4(afq[t], bb, num[j]);
            }
        }
        // phase 2: sc = tril(sigma_q @ sigma_k^T), tiles et <= wv
        for (int et = 0; et <= wv; ++et) {
            v8f acc = vzero;
            #pragma unroll
            for (int t = 0; t < 16; ++t) {
                v2f bb = load_b_bf16s(skL + et * 16 * 64, 1, 64, t, lane);
                acc = wmma_f32_k4(afq[t], bb, acc);
            }
            #pragma unroll
            for (int i = 0; i < 8; ++i) {
                int row = i + hi;
                float vsc = acc[i];
                if (et == wv && row < col) vsc = 0.f;
                sc[(wv * 16 + row) * CHK + et * 16 + col] = vsc;
            }
        }
        __syncthreads();

        // phase 3: den[c] = sigma_q[c].z + row-sum of sc
        if (tid < CHK) {
            int c = tid;
            float s = 0.f;
            #pragma unroll 8
            for (int d = 0; d < 64; ++d) s += sq[c * 64 + d] * zz[d];
            int emax = ((c >> 4) + 1) * 16;
            for (int e = 0; e < emax; ++e) s += sc[c * CHK + e];
            den[c] = s;
        }
        __syncthreads();

        // phase 4: num += sc @ v
        for (int et = 0; et <= wv; ++et) {
            #pragma unroll
            for (int t = 0; t < 4; ++t) {
                v2f a = load_a_f32(sc + (wv * 16) * CHK + et * 16, CHK, 1, t, lane);
                #pragma unroll
                for (int j = 0; j < 4; ++j) {
                    v2f bb = load_b_bf16s(svL + et * 16 * 64 + j * 16, 64, 1, t, lane);
                    num[j] = wmma_f32_k4(a, bb, num[j]);
                }
            }
        }
        // phase 5: out = g*mem + (1-g)*num/max(den,eps) -> bf16 (b,s,h*64+d)
        #pragma unroll
        for (int j = 0; j < 4; ++j)
            #pragma unroll
            for (int i = 0; i < 8; ++i) {
                int row = i + hi, c = wv * 16 + row, d = j * 16 + col;
                int s = s0 + c;
                float local = num[j][i] / fmaxf(den[c], EPS_F);
                float mo = memout[base + (size_t)s * 64 + d];
                float ov = g * mo + (1.f - g) * local;
                Ybf[((size_t)(b * SNUM + s)) * HID + h * 64 + d] = f32_to_bf16r(ov);
            }
        // phase 6: Skv += sigma_k^T @ v ; z += col-sums(sigma_k)
        #pragma unroll
        for (int tt = 0; tt < 2; ++tt) {
            int tile = wv * 2 + tt, di = tile >> 2, dj = tile & 3;
            v8f acc = vzero;
            #pragma unroll
            for (int t = 0; t < 32; ++t) {
                v2f a  = load_a_bf16s(skL + di * 16, 1, 64, t, lane);
                v2f bb = load_b_bf16s(svL + dj * 16, 64, 1, t, lane);
                acc = wmma_f32_k4(a, bb, acc);
            }
            #pragma unroll
            for (int i = 0; i < 8; ++i)
                Skv[(di * 16 + i + hi) * 64 + dj * 16 + col] += acc[i];
        }
        if (tid < 64) {
            float s = 0.f;
            for (int c = 0; c < CHK; ++c) s += bf16r_to_f32(skL[c * 64 + tid]);
            zz[tid] += s;
        }
        __syncthreads();
    }
}

// ---------------- host orchestration ----------------
extern "C" void kernel_launch(void* const* d_in, const int* in_sizes, int n_in,
                              void* d_out, int out_size, void* d_ws, size_t ws_size,
                              hipStream_t stream) {
    const float* hs     = (const float*)d_in[0];
    const float* w_q    = (const float*)d_in[1];
    const float* w_k    = (const float*)d_in[2];
    const float* w_v    = (const float*)d_in[3];
    const float* w_slc  = (const float*)d_in[4];
    const float* w_o    = (const float*)d_in[5];
    const float* gate   = (const float*)d_in[6];
    const float* mems   = (const float*)d_in[7];
    const float* mnorms = (const float*)d_in[8];
    const float* kseqs  = (const float*)d_in[9];
    const float* e_cls  = (const float*)d_in[10];
    const float* e_inw  = (const float*)d_in[11];
    const float* e_inb  = (const float*)d_in[12];
    const float* e_aow  = (const float*)d_in[13];
    const float* e_aob  = (const float*)d_in[14];
    const float* e_l1g  = (const float*)d_in[15];
    const float* e_l1b  = (const float*)d_in[16];
    const float* e_l2g  = (const float*)d_in[17];
    const float* e_l2b  = (const float*)d_in[18];
    const float* e_f1w  = (const float*)d_in[19];
    const float* e_f1b  = (const float*)d_in[20];
    const float* e_f2w  = (const float*)d_in[21];
    const float* e_f2b  = (const float*)d_in[22];
    const float* e_prj  = (const float*)d_in[23];

    char* ws = (char*)d_ws;
    size_t off = 0;
    auto alloc = [&](size_t bytes) -> void* {
        off = (off + 255) & ~(size_t)255;
        void* p = ws + off;
        off += bytes;
        return p;
    };
    const long NX = (long)BNUM * SNUM * HID;
    const long NW = (long)HID * HID;
    const long NQ = (long)BNUM * HNUM * SNUM * DNUM;

    unsigned short* Xbf  = (unsigned short*)alloc(NX * 2);
    unsigned short* Wqb  = (unsigned short*)alloc(NW * 2);
    unsigned short* Wkb  = (unsigned short*)alloc(NW * 2);
    unsigned short* Wvb  = (unsigned short*)alloc(NW * 2);
    unsigned short* Wsb  = (unsigned short*)alloc(NW * 2);
    unsigned short* Wob  = (unsigned short*)alloc(NW * 2);
    float* qbuf    = (float*)alloc(NQ * 4);
    float* kbuf    = (float*)alloc(NQ * 4);
    float* vbuf    = (float*)alloc(NQ * 4);
    float* qslcbuf = (float*)alloc(NQ * 4);
    float* memout  = (float*)alloc(NQ * 4);
    unsigned short* Ybf = (unsigned short*)alloc(NX * 2);
    float* landm   = (float*)alloc((size_t)MNUM * HNUM * DNUM * 4);
    float* wtsbuf  = (float*)alloc((size_t)MNUM * BNUM * HNUM * 4);

    cvt_f32_bf16_kernel<<<8192, 256, 0, stream>>>(hs, Xbf, NX);
    cvt_f32_bf16_kernel<<<2048, 256, 0, stream>>>(w_q,   Wqb, NW);
    cvt_f32_bf16_kernel<<<2048, 256, 0, stream>>>(w_k,   Wkb, NW);
    cvt_f32_bf16_kernel<<<2048, 256, 0, stream>>>(w_v,   Wvb, NW);
    cvt_f32_bf16_kernel<<<2048, 256, 0, stream>>>(w_slc, Wsb, NW);
    cvt_f32_bf16_kernel<<<2048, 256, 0, stream>>>(w_o,   Wob, NW);

    const int Mrows = BNUM * SNUM;
    const int gemm_blocks = (Mrows / 16) * (HID / 64) / 8;   // 1024
    gemm_bf16_kernel<<<gemm_blocks, 256, 0, stream>>>(Xbf, Wqb, qbuf,    Mrows, HID, HID, 1);
    gemm_bf16_kernel<<<gemm_blocks, 256, 0, stream>>>(Xbf, Wkb, kbuf,    Mrows, HID, HID, 1);
    gemm_bf16_kernel<<<gemm_blocks, 256, 0, stream>>>(Xbf, Wvb, vbuf,    Mrows, HID, HID, 1);
    gemm_bf16_kernel<<<gemm_blocks, 256, 0, stream>>>(Xbf, Wsb, qslcbuf, Mrows, HID, HID, 1);

    encoder_kernel<<<HNUM * MNUM, 256, 0, stream>>>(kseqs, e_cls, e_inw, e_inb, e_aow, e_aob,
                                                    e_l1g, e_l1b, e_l2g, e_l2b,
                                                    e_f1w, e_f1b, e_f2w, e_f2b, e_prj, landm);
    landmark_wts_kernel<<<BNUM * HNUM, 256, 0, stream>>>(qslcbuf, landm, wtsbuf);
    mem_readout_kernel<<<(BNUM * HNUM * SNUM / 16) / 8, 256, 0, stream>>>(qbuf, mems, mnorms, wtsbuf, memout);
    causal_kernel<<<BNUM * HNUM, 256, 0, stream>>>(qbuf, kbuf, vbuf, memout, gate, Ybf);
    gemm_bf16_kernel<<<gemm_blocks, 256, 0, stream>>>(Ybf, Wob, (float*)d_out, Mrows, HID, HID, 0);

    (void)in_sizes; (void)n_in; (void)out_size; (void)ws_size;
}
